// VisionLanguageAttention_42039139893700
// MI455X (gfx1250) — matrix operand order repaired and checked by reading, not compiled, not run on hardware
//
#include <hip/hip_runtime.h>
#include <hip/hip_bf16.h>

// ---------------------------------------------------------------------------
// VisionLanguageAttention for MI455X (gfx1250, wave32, WMMA bf16)
//
// q = lang @ Wq^T ; k = Wk @ vis ; v = Wv @ vis (per batch)
// att = softmax(q k / sqrt(d)) ; out = (att v) @ Wo^T
//
// All GEMMs run on v_wmma_f32_16x16x32_bf16. K/V projection is fused into a
// flash-attention kernel (one workgroup per (head,batch)): the 64MB-class
// fp32 K/V intermediates never touch HBM, and the 134MB vision tensor stays
// hot in the 192MB L2 across the 8 heads that re-read each batch slice.
// V is computed TRANSPOSED (Vt = vis^T * Wv^T) so both K and Vt accumulator
// tiles store to LDS with packed ds_store_b128 (r is contiguous along M in
// the WMMA C/D layout).
// ---------------------------------------------------------------------------

typedef __bf16 bhalf;
typedef __attribute__((ext_vector_type(16))) __bf16 v16bf;
typedef __attribute__((ext_vector_type(8)))  __bf16 v8bf;
typedef __attribute__((ext_vector_type(2)))  __bf16 v2bf;
typedef __attribute__((ext_vector_type(8)))  float  v8f;

#define SCALE_ATT 0.125f   // 64^-0.5

__device__ __forceinline__ v8f zero8() {
  v8f z;
#pragma unroll
  for (int i = 0; i < 8; ++i) z[i] = 0.f;
  return z;
}

__device__ __forceinline__ v16bf cat16(v8bf lo, v8bf hi) {
  v16bf r;
#pragma unroll
  for (int i = 0; i < 8; ++i) { r[i] = lo[i]; r[i + 8] = hi[i]; }
  return r;
}

// A-fragment (16x32 bf16), matrix stored row-major [row][k], leading dim ld.
// Lane L holds row (L&15); elements 0..7 = K (8*(L>>4))+0..7,
// elements 8..15 = K 16+(8*(L>>4))+0..7.
__device__ __forceinline__ v16bf load_a_frag(const bhalf* p, int ld, int row0, int col0) {
  const int lane = threadIdx.x & 31;
  const bhalf* q = p + (size_t)(row0 + (lane & 15)) * ld + col0 + ((lane >> 4) << 3);
  return cat16(*(const v8bf*)q, *(const v8bf*)(q + 16));
}

// B-fragment (32x16 bf16), stored as Mat[n][k] row-major (i.e. B^T), ld = k-stride.
// Lane L holds column N=(L&15); elements 0..7 = K (8*(L>>4))+0..7,
// elements 8..15 = K 16+(8*(L>>4))+0..7.
__device__ __forceinline__ v16bf load_b_frag(const bhalf* p, int ld, int n0, int k0) {
  const int lane = threadIdx.x & 31;
  const bhalf* q = p + (size_t)(n0 + (lane & 15)) * ld + k0 + ((lane >> 4) << 3);
  return cat16(*(const v8bf*)q, *(const v8bf*)(q + 16));
}

__device__ __forceinline__ v8f wmma_bf16(v16bf a, v16bf b, v8f c) {
  return __builtin_amdgcn_wmma_f32_16x16x32_bf16(false, a, false, b, (short)0, c,
                                                 false, false);
}

// ---------------------------------------------------------------------------
// fp32 -> bf16 conversion
// ---------------------------------------------------------------------------
__global__ void cvt_f32_to_bf16(const float* __restrict__ src,
                                bhalf* __restrict__ dst, int n) {
  int i = blockIdx.x * blockDim.x + threadIdx.x;
  if (i < n) dst[i] = (bhalf)src[i];
}

// ---------------------------------------------------------------------------
// Out[M,N] = A[M,K] * W[N,K]^T   (bf16 in, f32 accumulate)
// 128 threads = 4 waves; wave owns a 16-row strip. grid = (M/64, N/128).
// One A-fragment feeds 4 WMMAs (4 n-tile accumulators live per group).
// ---------------------------------------------------------------------------
template <bool OUT_BF16>
__global__ __launch_bounds__(128) void gemm_abT(const bhalf* __restrict__ A,
                                                const bhalf* __restrict__ W,
                                                void* __restrict__ Out,
                                                int M, int N, int K) {
  const int lane = threadIdx.x & 31;
  const int wave = threadIdx.x >> 5;
  const int m0 = blockIdx.x * 64 + wave * 16;
  const int nbase = blockIdx.y * 128;
  if (m0 >= M) return;

  for (int ng = 0; ng < 2; ++ng) {
    const int n0 = nbase + ng * 64;
    v8f acc[4];
#pragma unroll
    for (int j = 0; j < 4; ++j) acc[j] = zero8();
    for (int kk = 0; kk < K; kk += 32) {
      const v16bf a = load_a_frag(A, K, m0, kk);
#pragma unroll
      for (int j = 0; j < 4; ++j)
        acc[j] = wmma_bf16(a, load_b_frag(W, K, n0 + j * 16, kk), acc[j]);
    }
#pragma unroll
    for (int j = 0; j < 4; ++j) {
      const int n = n0 + j * 16 + (lane & 15);
#pragma unroll
      for (int r = 0; r < 8; ++r) {
        const int m = m0 + r + 8 * (lane >> 4);
        if constexpr (OUT_BF16)
          ((bhalf*)Out)[(size_t)m * N + n] = (bhalf)acc[j][r];
        else
          ((float*)Out)[(size_t)m * N + n] = acc[j][r];
      }
    }
  }
}

// ---------------------------------------------------------------------------
// Fused K/V projection + flash attention.
// grid = (heads=8, batch=16); block = 128 threads = 4 waves.
// Wave w owns T-strip rows [16w,16w+16); in projection it owns the d-strip of
// K and the s-strip of Vt. S is processed in chunks of 64.
// ---------------------------------------------------------------------------
__global__ __launch_bounds__(128) void attn_fused(const float* __restrict__ vis,
                                                  const bhalf* __restrict__ qbf,
                                                  const bhalf* __restrict__ Wkbf,
                                                  const bhalf* __restrict__ Wvbf,
                                                  bhalf* __restrict__ outbf) {
  constexpr int C = 512, S = 4096, D = 64, T = 64, SC = 64;
  const int h = blockIdx.x;
  const int b = blockIdx.y;
  const int tid = threadIdx.x;
  const int wave = tid >> 5;
  const int lane = tid & 31;
  const int lhalf = lane >> 4;
  const int lmod = lane & 15;

  __shared__ bhalf ldsVis[64][64];  // vis tile, N(=s)-major: [s][c]   8 KB
  __shared__ bhalf ldsK[64][64];    // K chunk, N(=s)-major: [s][d]    8 KB
  __shared__ bhalf ldsV[64][64];    // V chunk, N(=d)-major: [d][s]    8 KB
  __shared__ bhalf ldsP[64][64];    // probabilities, [t][s]           8 KB

  const bhalf* qbase = qbf + (size_t)b * T * 512 + h * D;
  const bhalf* wkbase = Wkbf + (size_t)(h * D) * C;
  const bhalf* wvbase = Wvbf + (size_t)(h * D) * C;

  // Q fragments for this wave's T-strip (reused across all chunks)
  const v16bf qf0 = load_a_frag(qbase, 512, 16 * wave, 0);
  const v16bf qf1 = load_a_frag(qbase, 512, 16 * wave, 32);

  v8f acc_o[4];
  float mrow[8], lrow[8];
#pragma unroll
  for (int i = 0; i < 4; ++i) acc_o[i] = zero8();
#pragma unroll
  for (int r = 0; r < 8; ++r) { mrow[r] = -1.0e30f; lrow[r] = 0.f; }

  // staging coordinates: two adjacent c-rows per thread, 16 s each
  const int c0 = (tid >> 2) * 2;
  const int sc = (tid & 3) * 16;

  for (int chunk = 0; chunk < S / SC; ++chunk) {
    const int s0 = chunk * SC;

    // ---- Phase A: Kc[d][s] = Wk_h * vis ; Vt[s][d] = vis^T * Wv_h^T
    v8f acc_k[4], acc_vt[4];
#pragma unroll
    for (int i = 0; i < 4; ++i) { acc_k[i] = zero8(); acc_vt[i] = zero8(); }

    for (int st = 0; st < C / 64; ++st) {
      const int ck = st * 64;
      __syncthreads();  // ldsVis reuse
      {  // stage vis[ck..ck+63][s0..s0+63] fp32 -> bf16, transposed to [s][c]
        const float* r0 = vis + ((size_t)b * C + ck + c0) * S + s0 + sc;
        const float* r1 = r0 + S;
#pragma unroll
        for (int jj = 0; jj < 4; ++jj) {
          const float4 f0 = *(const float4*)(r0 + jj * 4);
          const float4 f1 = *(const float4*)(r1 + jj * 4);
          v2bf p;
          p[0] = (bhalf)f0.x; p[1] = (bhalf)f1.x; *(v2bf*)&ldsVis[sc + jj * 4 + 0][c0] = p;
          p[0] = (bhalf)f0.y; p[1] = (bhalf)f1.y; *(v2bf*)&ldsVis[sc + jj * 4 + 1][c0] = p;
          p[0] = (bhalf)f0.z; p[1] = (bhalf)f1.z; *(v2bf*)&ldsVis[sc + jj * 4 + 2][c0] = p;
          p[0] = (bhalf)f0.w; p[1] = (bhalf)f1.w; *(v2bf*)&ldsVis[sc + jj * 4 + 3][c0] = p;
        }
        if (st + 1 < C / 64) {  // prefetch next c-block (global_prefetch_b8)
          __builtin_prefetch(r0 + (size_t)64 * S, 0, 1);
          __builtin_prefetch(r1 + (size_t)64 * S, 0, 1);
        }
      }
      __syncthreads();

#pragma unroll
      for (int half = 0; half < 2; ++half) {
        const int k0 = half * 32;
        const v16bf ak = load_a_frag(wkbase, C, 16 * wave, ck + k0);           // Wk rows
        const v16bf avt = load_a_frag(&ldsVis[0][0], 64, 16 * wave, k0);       // vis^T rows
#pragma unroll
        for (int n = 0; n < 4; ++n) {
          acc_k[n] = wmma_bf16(ak, load_b_frag(&ldsVis[0][0], 64, 16 * n, k0), acc_k[n]);
          acc_vt[n] = wmma_bf16(avt, load_b_frag(wvbase, C, 16 * n, ck + k0), acc_vt[n]);
        }
      }
    }

    // packed spill: K tile (M=d strip) -> ldsK[s][d], Vt tile (M=s strip) ->
    // ldsV[d][s]; accumulator element r is contiguous along M => b128 stores
#pragma unroll
    for (int n = 0; n < 4; ++n) {
      v8bf pk, pv;
#pragma unroll
      for (int r = 0; r < 8; ++r) {
        pk[r] = (bhalf)acc_k[n][r];
        pv[r] = (bhalf)acc_vt[n][r];
      }
      *(v8bf*)&ldsK[16 * n + lmod][16 * wave + 8 * lhalf] = pk;
      *(v8bf*)&ldsV[16 * n + lmod][16 * wave + 8 * lhalf] = pv;
    }
    __syncthreads();

    // ---- Phase B: scores = Q * Kc
    v8f acc_s[4];
#pragma unroll
    for (int i = 0; i < 4; ++i) acc_s[i] = zero8();
#pragma unroll
    for (int n = 0; n < 4; ++n) {
      acc_s[n] = wmma_bf16(qf0, load_b_frag(&ldsK[0][0], 64, 16 * n, 0), acc_s[n]);
      acc_s[n] = wmma_bf16(qf1, load_b_frag(&ldsK[0][0], 64, 16 * n, 32), acc_s[n]);
    }

    // ---- online softmax (row reductions stay inside 16-lane halves)
    float mloc[8];
#pragma unroll
    for (int r = 0; r < 8; ++r) {
      float m = fmaxf(fmaxf(acc_s[0][r], acc_s[1][r]),
                      fmaxf(acc_s[2][r], acc_s[3][r]));
#pragma unroll
      for (int off = 8; off; off >>= 1) m = fmaxf(m, __shfl_xor(m, off, 32));
      mloc[r] = m * SCALE_ATT;
    }
    float cf[8];
#pragma unroll
    for (int r = 0; r < 8; ++r) {
      const float mn = fmaxf(mrow[r], mloc[r]);
      cf[r] = __expf(mrow[r] - mn);
      mrow[r] = mn;
    }
    float ps[8];
#pragma unroll
    for (int r = 0; r < 8; ++r) ps[r] = 0.f;
#pragma unroll
    for (int n = 0; n < 4; ++n) {
#pragma unroll
      for (int r = 0; r < 8; ++r) {
        const float p = __expf(acc_s[n][r] * SCALE_ATT - mrow[r]);
        ps[r] += p;
        ldsP[16 * wave + r + 8 * lhalf][16 * n + lmod] = (bhalf)p;
      }
    }
#pragma unroll
    for (int r = 0; r < 8; ++r) {
      float s = ps[r];
#pragma unroll
      for (int off = 8; off; off >>= 1) s += __shfl_xor(s, off, 32);
      lrow[r] = lrow[r] * cf[r] + s;
    }
#pragma unroll
    for (int nd = 0; nd < 4; ++nd)
#pragma unroll
      for (int r = 0; r < 8; ++r) acc_o[nd][r] *= cf[r];

    // ---- O += P * V (ldsP rows are wave-private, ldsV synced above)
    const v16bf p0 = load_a_frag(&ldsP[0][0], 64, 16 * wave, 0);
    const v16bf p1 = load_a_frag(&ldsP[0][0], 64, 16 * wave, 32);
#pragma unroll
    for (int nd = 0; nd < 4; ++nd) {
      acc_o[nd] = wmma_bf16(p0, load_b_frag(&ldsV[0][0], 64, 16 * nd, 0), acc_o[nd]);
      acc_o[nd] = wmma_bf16(p1, load_b_frag(&ldsV[0][0], 64, 16 * nd, 32), acc_o[nd]);
    }
  }

  // ---- epilogue: normalize and store (B, T, h*64+d) in bf16
#pragma unroll
  for (int r = 0; r < 8; ++r) {
    const float inv = 1.f / lrow[r];
    const int t = 16 * wave + r + 8 * lhalf;
#pragma unroll
    for (int nd = 0; nd < 4; ++nd) {
      const int d = 16 * nd + lmod;
      outbf[((size_t)b * T + t) * 512 + h * D + d] = (bhalf)(acc_o[nd][r] * inv);
    }
  }
}

// ---------------------------------------------------------------------------
// launch
// ---------------------------------------------------------------------------
extern "C" void kernel_launch(void* const* d_in, const int* in_sizes, int n_in,
                              void* d_out, int out_size, void* d_ws, size_t ws_size,
                              hipStream_t stream) {
  const float* vis  = (const float*)d_in[0];  // (16, 512, 64, 64)
  const float* lang = (const float*)d_in[1];  // (16, 64, 512)
  // d_in[2]: mask (unused by forward)
  const float* Wq = (const float*)d_in[3];    // (512, 512)
  const float* Wk = (const float*)d_in[4];
  const float* Wv = (const float*)d_in[5];
  const float* Wo = (const float*)d_in[6];

  constexpr int LANG_N = 16 * 64 * 512;  // 524288
  constexpr int W_N    = 512 * 512;      // 262144

  bhalf* ws      = (bhalf*)d_ws;
  bhalf* lang_bf = ws;
  bhalf* Wq_bf   = lang_bf + LANG_N;
  bhalf* Wk_bf   = Wq_bf + W_N;
  bhalf* Wv_bf   = Wk_bf + W_N;
  bhalf* Wo_bf   = Wv_bf + W_N;
  bhalf* q_bf    = Wo_bf + W_N;          // (B*T, 512)
  bhalf* attn_bf = q_bf + LANG_N;        // (B*T, 512)

  cvt_f32_to_bf16<<<(LANG_N + 255) / 256, 256, 0, stream>>>(lang, lang_bf, LANG_N);
  cvt_f32_to_bf16<<<(W_N + 255) / 256, 256, 0, stream>>>(Wq, Wq_bf, W_N);
  cvt_f32_to_bf16<<<(W_N + 255) / 256, 256, 0, stream>>>(Wk, Wk_bf, W_N);
  cvt_f32_to_bf16<<<(W_N + 255) / 256, 256, 0, stream>>>(Wv, Wv_bf, W_N);
  cvt_f32_to_bf16<<<(W_N + 255) / 256, 256, 0, stream>>>(Wo, Wo_bf, W_N);

  // q = lang @ Wq^T  (M=1024, N=512, K=512), bf16 out
  gemm_abT<true><<<dim3(16, 4), 128, 0, stream>>>(lang_bf, Wq_bf, q_bf, 1024, 512, 512);

  // fused K/V projection + flash attention
  attn_fused<<<dim3(8, 16), 128, 0, stream>>>(vis, q_bf, Wk_bf, Wv_bf, attn_bf);

  // out = attn @ Wo^T (M=1024, N=512, K=512), fp32 out
  gemm_abT<false><<<dim3(16, 4), 128, 0, stream>>>(attn_bf, Wo_bf, d_out, 1024, 512, 512);
}
